// MinRNN_66168266162555
// MI455X (gfx1250) — matrix operand order repaired and checked by reading, not compiled, not run on hardware
//
#include <hip/hip_runtime.h>
#include <hip/hip_bf16.h>

#define BB 8
#define SS 512
#define UU 2560
#define EE 1024
#define VV 32000
#define MM (BB * SS)   // 4096 rows of x
#define MBLK 4         // M-tiles per wave (64 rows)

typedef __attribute__((ext_vector_type(16))) __bf16 bf16x16;
typedef __attribute__((ext_vector_type(8)))  __bf16 bf16x8;
typedef __attribute__((ext_vector_type(4)))  __bf16 bf16x4;
typedef __attribute__((ext_vector_type(8)))  float  f32x8;

static __device__ __forceinline__ __bf16 f2bf(float f) {
    union { float f; unsigned u; } x; x.f = f;
    unsigned r = x.u + 0x7FFFu + ((x.u >> 16) & 1u);   // round-to-nearest-even
    unsigned short h = (unsigned short)(r >> 16);
    return __builtin_bit_cast(__bf16, h);
}

// ---------------- Stage 1a: gather embedding rows -> bf16 x ----------------
__global__ void gather_embed_bf16(const int* __restrict__ sent,
                                  const float* __restrict__ emb,
                                  __bf16* __restrict__ xb) {
    int m   = blockIdx.x;
    int tok = sent[m];
    const float4* src = (const float4*)(emb + (size_t)tok * EE);
    float4 fv = src[threadIdx.x];
    bf16x4 o;
    o[0] = f2bf(fv.x); o[1] = f2bf(fv.y); o[2] = f2bf(fv.z); o[3] = f2bf(fv.w);
    *(bf16x4*)(xb + (size_t)m * EE + (size_t)threadIdx.x * 4) = o;
}

// ---------------- Stage 1b: fp32 -> bf16 weight convert ----------------
__global__ void convert_f32_bf16(const float* __restrict__ src,
                                 __bf16* __restrict__ dst, int n4) {
    int i = blockIdx.x * blockDim.x + threadIdx.x;
    if (i >= n4) return;
    float4 fv = ((const float4*)src)[i];
    bf16x4 o;
    o[0] = f2bf(fv.x); o[1] = f2bf(fv.y); o[2] = f2bf(fv.z); o[3] = f2bf(fv.w);
    *(bf16x4*)(dst + (size_t)i * 4) = o;
}

// ---------------- Stage 2: fused WMMA gates GEMM + elementwise ----------------
// Each wave owns a 64x16 output tile (MBLK=4 m-subtiles) for all 3 gates.
// Per k-step: 3 B fragments + 4 A fragments -> 12 WMMAs (B reused across M).
// grid = (MM/(16*MBLK), UU/(16*8)), block = 256 threads (8 waves).
__global__ void __launch_bounds__(256)
gates_wmma(const __bf16* __restrict__ xb,
           const __bf16* __restrict__ wf,
           const __bf16* __restrict__ wi,
           const __bf16* __restrict__ wh,
           const float* __restrict__ bfv_,
           const float* __restrict__ biv_,
           const float* __restrict__ bhv_,
           float* __restrict__ aArr,
           float* __restrict__ vArr) {
    const int wave = threadIdx.x >> 5;
    const int lane = threadIdx.x & 31;
    const int half = lane >> 4;     // 0: lanes 0-15, 1: lanes 16-31
    const int l    = lane & 15;

    const int m0 = blockIdx.x * (16 * MBLK);
    const int u0 = (blockIdx.y * 8 + wave) * 16;

    f32x8 accF[MBLK], accI[MBLK], accH[MBLK];
#pragma unroll
    for (int mb = 0; mb < MBLK; ++mb) { accF[mb] = {}; accI[mb] = {}; accH[mb] = {}; }

    // A: per ISA layout, lane holds row m0 + mb*16 + l; this code's K order (not ISA table order): k = 16*half + s, lo/hi by lane half
    const __bf16* xrow  = xb + (size_t)(m0 + l) * EE;           // mb offset folds into ioffset
    // B: column u0+l (= weight row u0+l); lanes 16-31 take K offset +16
    const __bf16* wfrow = wf + (size_t)(u0 + l) * EE + half * 16;
    const __bf16* wirow = wi + (size_t)(u0 + l) * EE + half * 16;
    const __bf16* whrow = wh + (size_t)(u0 + l) * EE + half * 16;

    for (int k0 = 0; k0 < EE; k0 += 32) {
        bf16x16 bF = *(const bf16x16*)(wfrow + k0);
        bf16x16 bI = *(const bf16x16*)(wirow + k0);
        bf16x16 bH = *(const bf16x16*)(whrow + k0);

#pragma unroll
        for (int mb = 0; mb < MBLK; ++mb) {
            const __bf16* xr = xrow + (size_t)mb * 16 * EE;
            bf16x8 alo = *(const bf16x8*)(xr + k0 + half * 16);
            bf16x8 ahi = *(const bf16x8*)(xr + k0 + half * 16 + 8);
            bf16x16 afrag;
#pragma unroll
            for (int j = 0; j < 8; ++j) { afrag[j] = alo[j]; afrag[j + 8] = ahi[j]; }

            accF[mb] = __builtin_amdgcn_wmma_f32_16x16x32_bf16(
                false, afrag, false, bF, (short)0, accF[mb], false, false);
            accI[mb] = __builtin_amdgcn_wmma_f32_16x16x32_bf16(
                false, afrag, false, bI, (short)0, accI[mb], false, false);
            accH[mb] = __builtin_amdgcn_wmma_f32_16x16x32_bf16(
                false, afrag, false, bH, (short)0, accH[mb], false, false);
        }
    }

    // Epilogue: C/D layout -> element r is (row mbase + r, col u0 + l)
    const int col   = u0 + l;
    const float bfb = bfv_[col];
    const float bib = biv_[col];
    const float bhb = bhv_[col];
#pragma unroll
    for (int mb = 0; mb < MBLK; ++mb) {
        const int mbase = m0 + mb * 16 + half * 8;
#pragma unroll
        for (int r = 0; r < 8; ++r) {
            float fo = 1.0f / (1.0f + __expf(-(accF[mb][r] + bfb)));
            float io = 1.0f / (1.0f + __expf(-(accI[mb][r] + bib)));
            float ht = accH[mb][r] + bhb;
            float s  = fo + io;
            size_t off = (size_t)(mbase + r) * UU + col;
            aArr[off] = fo / s;
            vArr[off] = (io / s) * ht;
        }
    }
}

// ---------------- Stage 3: sequential scan over time, parallel over channels ----
__global__ void scan_channels(const float* __restrict__ aArr,
                              const float* __restrict__ vArr,
                              float* __restrict__ hT) {
    int c = blockIdx.x * blockDim.x + threadIdx.x;   // channel in [0, B*U)
    if (c >= BB * UU) return;
    int b = c / UU;
    int u = c - b * UU;
    const float* ap = aArr + (size_t)b * SS * UU + u;
    const float* vp = vArr + (size_t)b * SS * UU + u;
    float h = 0.0f;
    for (int t = 0; t < SS; ++t) {
        size_t o = (size_t)t * UU;
        h = fmaf(ap[o], h, vp[o]);
    }
    hT[c] = h;   // hT flat == (B, U) row-major == reshape(B*S, 5) flat
}

// ---------------- Stage 4: (4096,5) x (5,32000) + bc, store-bound ----------------
__global__ void out_gemm(const float* __restrict__ hT,
                         const float* __restrict__ Wc,
                         const float* __restrict__ bc,
                         float* __restrict__ out) {
    __shared__ float hs[5];
    int r = blockIdx.y;
    if (threadIdx.x < 5) hs[threadIdx.x] = hT[(size_t)r * 5 + threadIdx.x];
    __syncthreads();
    int v4 = (blockIdx.x * blockDim.x + threadIdx.x) * 4;
    if (v4 >= VV) return;              // V % 4 == 0, whole quad valid or not
    float4 bcv = *(const float4*)(bc + v4);
    float acc[4] = {bcv.x, bcv.y, bcv.z, bcv.w};
#pragma unroll
    for (int q = 0; q < 4; ++q) {
        const float* wr = Wc + (size_t)(v4 + q) * 5;
#pragma unroll
        for (int j = 0; j < 5; ++j) acc[q] = fmaf(hs[j], wr[j], acc[q]);
    }
    float4 o = {acc[0], acc[1], acc[2], acc[3]};
    *(float4*)(out + (size_t)r * VV + v4) = o;
}

extern "C" void kernel_launch(void* const* d_in, const int* in_sizes, int n_in,
                              void* d_out, int out_size, void* d_ws, size_t ws_size,
                              hipStream_t stream) {
    const int*   sent = (const int*)  d_in[0];
    const float* emb  = (const float*)d_in[1];
    const float* Wf   = (const float*)d_in[2];
    const float* bf_  = (const float*)d_in[3];
    const float* Wi   = (const float*)d_in[4];
    const float* bi_  = (const float*)d_in[5];
    const float* Wh   = (const float*)d_in[6];
    const float* bh_  = (const float*)d_in[7];
    const float* Wc   = (const float*)d_in[8];
    const float* bc   = (const float*)d_in[9];
    float* out = (float*)d_out;

    // Workspace layout (bytes)
    char* ws = (char*)d_ws;
    const size_t xb_off = 0;
    const size_t xb_sz  = (size_t)MM * EE * 2;                 // 8 MB
    const size_t w_sz   = (size_t)UU * EE * 2;                 // 5.25 MB each
    const size_t wf_off = xb_off + xb_sz;
    const size_t wi_off = wf_off + w_sz;
    const size_t wh_off = wi_off + w_sz;
    const size_t a_off  = wh_off + w_sz;
    const size_t av_sz  = (size_t)MM * UU * 4;                 // 41.9 MB each
    const size_t v_off  = a_off + av_sz;
    const size_t h_off  = v_off + av_sz;

    __bf16* xb  = (__bf16*)(ws + xb_off);
    __bf16* wfb = (__bf16*)(ws + wf_off);
    __bf16* wib = (__bf16*)(ws + wi_off);
    __bf16* whb = (__bf16*)(ws + wh_off);
    float*  aArr = (float*)(ws + a_off);
    float*  vArr = (float*)(ws + v_off);
    float*  hT   = (float*)(ws + h_off);

    // Stage 1: precision convert + gather
    gather_embed_bf16<<<MM, 256, 0, stream>>>(sent, emb, xb);
    {
        int n4 = (UU * EE) / 4;                                 // 655360
        int blocks = (n4 + 255) / 256;
        convert_f32_bf16<<<blocks, 256, 0, stream>>>(Wf, wfb, n4);
        convert_f32_bf16<<<blocks, 256, 0, stream>>>(Wi, wib, n4);
        convert_f32_bf16<<<blocks, 256, 0, stream>>>(Wh, whb, n4);
    }

    // Stage 2: fused gates WMMA GEMM -> a, v   (64x16 tile per wave)
    {
        dim3 grid(MM / (16 * MBLK), UU / (16 * 8));             // (64, 20)
        gates_wmma<<<grid, 256, 0, stream>>>(xb, wfb, wib, whb,
                                             bf_, bi_, bh_, aArr, vArr);
    }

    // Stage 3: recurrence scan
    {
        int n = BB * UU;                                        // 20480
        scan_channels<<<(n + 255) / 256, 256, 0, stream>>>(aArr, vArr, hT);
    }

    // Stage 4: output projection (store-bound)
    {
        dim3 grid((VV + 1023) / 1024, MM);                      // (32, 4096)
        out_gemm<<<grid, 256, 0, stream>>>(hT, Wc, bc, out);
    }
    (void)in_sizes; (void)n_in; (void)out_size; (void)ws_size;
}
